// OneTransBackbone_52321291600428
// MI455X (gfx1250) — matrix operand order repaired and checked by reading, not compiled, not run on hardware
//
#include <hip/hip_runtime.h>
#include <hip/hip_bf16.h>
#include <math.h>

// ---------------- problem constants ----------------
constexpr int B_  = 16;
constexpr int L0_ = 512;
constexpr int L1_ = 256;
constexpr int NS_ = 32;
constexpr int D_  = 256;
constexpr int H_  = 8;
constexpr int NL_ = 4;
constexpr int HM_ = 4;
constexpr int T_  = L0_ + L1_ + NS_ + 1;   // 801
constexpr int HD_ = D_ / H_;               // 32
constexpr int M_  = B_ * T_;               // 12816
constexpr int TP_ = 832;                   // padded key length (mult of 32)
constexpr int KT_ = TP_ / 16;              // 52 key tiles
constexpr int QT_ = (T_ + 15) / 16;        // 51 query tiles
constexpr int MT32_ = (M_ + 31) / 32;      // 401 32-row tiles
constexpr int KPANEL_ = 512;               // A-panel K chunk staged in LDS
constexpr float SCALE_ = 0.1767766952966369f;
constexpr float EPS_ = 1e-5f;

// ---------------- WMMA plumbing ----------------
typedef __attribute__((ext_vector_type(16))) __bf16 v16bf;
typedef __attribute__((ext_vector_type(8)))  float  v8f;
typedef __attribute__((ext_vector_type(4)))  unsigned int u32x4;   // plain vector (no ctors)
typedef __attribute__((ext_vector_type(8)))  int          tdm_v8i;
typedef __attribute__((ext_vector_type(4)))  int          tdm_v4i;
typedef __attribute__((ext_vector_type(4)))  unsigned int tdm_v4u;

// explicit address-space pointers: force global_load_b128 / ds_load_b128
typedef const u32x4 __attribute__((address_space(1)))* g4p;
typedef const u32x4 __attribute__((address_space(3)))* l4p;

#if defined(__has_builtin)
#  if __has_builtin(__builtin_amdgcn_tensor_load_to_lds)
#    define HAVE_TDM 1
#  endif
#endif

__device__ __forceinline__ v8f wmma_bf16(v16bf a, v16bf b, v8f c) {
    return __builtin_amdgcn_wmma_f32_16x16x32_bf16(
        false, a, false, b, (short)0, c, false, false);
}

__device__ __forceinline__ v16bf frag_from(u32x4 lo_, u32x4 hi_) {
    union U { u32x4 q; __bf16 h[8]; } a, b;
    a.q = lo_; b.q = hi_;
    v16bf f;
#pragma unroll
    for (int i = 0; i < 8; ++i) { f[i] = a.h[i]; f[i + 8] = b.h[i]; }
    return f;
}
// A pattern: elems 0..7 at p, elems 8..15 at p+16 elems (interleaved K halves).
__device__ __forceinline__ v16bf load_a_g(const __bf16* p) {
    g4p q = (g4p)(unsigned long long)(size_t)p;
    return frag_from(q[0], q[2]);
}
__device__ __forceinline__ v16bf load_a_lds(const __bf16* p) {
    l4p q = (l4p)(size_t)(unsigned)(size_t)p;   // low 32 bits = LDS offset
    return frag_from(q[0], q[2]);
}
// B pattern: elems 0..7 at p, elems 8..15 at p+16 elems (interleaved K halves).
__device__ __forceinline__ v16bf load_b_g(const __bf16* p) {
    g4p q = (g4p)(unsigned long long)(size_t)p;
    return frag_from(q[0], q[2]);
}
// Per-lane base offsets.
__device__ __forceinline__ const __bf16* a_lane_base(const __bf16* tile, int ld) {
    int lane = threadIdx.x & 31;
    return tile + (size_t)(lane & 15) * ld + ((lane >> 4) << 3);
}
__device__ __forceinline__ const __bf16* b_lane_base(const __bf16* tile, int ld) {
    int lane = threadIdx.x & 31;
    return tile + (size_t)(lane & 15) * ld + ((lane >> 4) << 3);
}

// ---------------- TDM: stage a 2D bf16 panel into LDS ----------------
__device__ __forceinline__ void tdm_load_panel(__bf16* ldsDst, const __bf16* src,
                                               int rowElems, int rows, int strideElems)
{
#ifdef HAVE_TDM
    unsigned lds = (unsigned)(size_t)(void*)ldsDst;     // low 32 bits = LDS offset
    unsigned long long ga = (unsigned long long)(size_t)src;
    unsigned d0 = (unsigned)rowElems, d1 = (unsigned)rows;
    unsigned long long s0 = (unsigned long long)strideElems;
    tdm_v4u g0;
    g0[0] = 1u;                                         // count=1, user descriptor
    g0[1] = lds;                                        // lds_addr
    g0[2] = (unsigned)(ga & 0xffffffffu);               // global_addr[31:0]
    g0[3] = (unsigned)((ga >> 32) & 0x01ffffffu) | (2u << 30); // addr[56:32], type=2
    tdm_v8i g1;
    g1[0] = (int)(1u << 16);                            // wg_mask=0, data_size=1 (2B)
    g1[1] = (int)((d0 & 0xffffu) << 16);                // tensor_dim0 lo
    g1[2] = (int)(((d0 >> 16) & 0xffffu) | ((d1 & 0xffffu) << 16));
    g1[3] = (int)(((d1 >> 16) & 0xffffu) | ((d0 & 0xffffu) << 16)); // tile_dim0
    g1[4] = (int)(d1 & 0xffffu);                        // tile_dim1, tile_dim2=0
    g1[5] = (int)(s0 & 0xffffffffu);                    // dim0_stride lo
    g1[6] = (int)((s0 >> 32) & 0xffffu);                // dim0_stride hi
    g1[7] = 0;
    tdm_v4i g2 = {0, 0, 0, 0};
    tdm_v4i g3 = {0, 0, 0, 0};
#  if __clang_major__ >= 23
    tdm_v8i g1b = {0, 0, 0, 0, 0, 0, 0, 0};
    __builtin_amdgcn_tensor_load_to_lds(g0, g1, g2, g3, g1b, 0);
#  else
    __builtin_amdgcn_tensor_load_to_lds(g0, g1, g2, g3, 0);
#  endif
    __builtin_amdgcn_s_wait_tensorcnt(0);
#else
    for (int r = 0; r < rows; ++r) {
        const u32x4* s = (const u32x4*)(src + (size_t)r * strideElems);
        u32x4* d = (u32x4*)(ldsDst + (size_t)r * rowElems);
        for (int i = (int)(threadIdx.x & 31); i < rowElems / 8; i += 32) d[i] = s[i];
    }
#endif
}

// ---------------- elementwise / prep kernels ----------------
__global__ __launch_bounds__(256) void embed_k(
    const float* __restrict__ s0, const float* __restrict__ s1,
    const float* __restrict__ ns, const float* __restrict__ dom,
    const float* __restrict__ cls, float* __restrict__ X)
{
    int t = blockIdx.x, b = blockIdx.y, c = threadIdx.x;
    float v;
    if (t < L0_)                 v = s0[((size_t)b * L0_ + t) * D_ + c] + dom[c];
    else if (t < L0_ + L1_)      v = s1[((size_t)b * L1_ + (t - L0_)) * D_ + c] + dom[D_ + c];
    else if (t < L0_ + L1_ + NS_) v = ns[((size_t)b * NS_ + (t - L0_ - L1_)) * D_ + c];
    else                         v = cls[c];
    X[((size_t)b * T_ + t) * D_ + c] = v;
}

__global__ __launch_bounds__(256) void prep_wt_k(
    const float* __restrict__ W, __bf16* __restrict__ Wt, int K, int N)
{
    int idx = blockIdx.x * blockDim.x + threadIdx.x;
    if (idx >= K * N) return;
    int n = idx / K, k = idx - n * K;
    Wt[idx] = (__bf16)W[(size_t)k * N + n];
}

__global__ __launch_bounds__(256) void ln_bf16_k(
    const float* __restrict__ X, const float* __restrict__ w,
    const float* __restrict__ bb, __bf16* __restrict__ out)
{
    int m = blockIdx.x, c = threadIdx.x;
    __shared__ float s1[256], s2[256];
    float v = X[(size_t)m * D_ + c];
    s1[c] = v; s2[c] = v * v;
    __syncthreads();
    for (int st = 128; st > 0; st >>= 1) {
        if (c < st) { s1[c] += s1[c + st]; s2[c] += s2[c + st]; }
        __syncthreads();
    }
    float mean = s1[0] * (1.f / D_);
    float var  = s2[0] * (1.f / D_) - mean * mean;
    float r = rsqrtf(var + EPS_);
    out[(size_t)m * D_ + c] = (__bf16)((v - mean) * r * w[c] + bb[c]);
}

__global__ __launch_bounds__(256) void final_ln_k(
    const float* __restrict__ X, const float* __restrict__ w,
    const float* __restrict__ bb, float* __restrict__ out)
{
    int b = blockIdx.x, c = threadIdx.x;
    size_t m = (size_t)b * T_ + (T_ - 1);
    __shared__ float s1[256], s2[256];
    float v = X[m * D_ + c];
    s1[c] = v; s2[c] = v * v;
    __syncthreads();
    for (int st = 128; st > 0; st >>= 1) {
        if (c < st) { s1[c] += s1[c + st]; s2[c] += s2[c + st]; }
        __syncthreads();
    }
    float mean = s1[0] * (1.f / D_);
    float var  = s2[0] * (1.f / D_) - mean * mean;
    float r = rsqrtf(var + EPS_);
    out[(size_t)b * D_ + c] = (v - mean) * r * w[c] + bb[c];
}

// ---------------- generic bf16 WMMA GEMM ----------------
// Block: 128 threads = 4 waves; block tile 32M x 256N; wave tile 32M x 64N.
__global__ __launch_bounds__(128) void gemm_bf16_k(
    const __bf16* __restrict__ A, const __bf16* __restrict__ Wt,
    const float* __restrict__ bias, int M, int N, int K, int op,
    __bf16* __restrict__ outh, float* __restrict__ resid)
{
    __shared__ __align__(16) __bf16 sA[32 * KPANEL_];   // 32 KB
    int wave = threadIdx.x >> 5, lane = threadIdx.x & 31;
    int m0 = blockIdx.x * 32;
    int n0 = blockIdx.y * 256 + wave * 64;
    int rowsA = (M - m0 >= 32) ? 32 : (M - m0);

    v8f acc[2][4];
#pragma unroll
    for (int mt = 0; mt < 2; ++mt)
#pragma unroll
        for (int j = 0; j < 4; ++j) acc[mt][j] = v8f{};

    const __bf16* pB[4];
#pragma unroll
    for (int j = 0; j < 4; ++j)
        pB[j] = b_lane_base(Wt + (size_t)(n0 + j * 16) * K, K);

    for (int kp = 0; kp < K; kp += KPANEL_) {
        int kw = (K - kp >= KPANEL_) ? KPANEL_ : (K - kp);
        if (wave == 0) tdm_load_panel(sA, A + (size_t)m0 * K + kp, kw, rowsA, K);
        __syncthreads();

        const __bf16* pA0 = a_lane_base(sA, kw);
        const __bf16* pA1 = pA0 + (size_t)16 * kw;
        v16bf b0[4], b1[4];
#pragma unroll
        for (int j = 0; j < 4; ++j) b0[j] = load_b_g(pB[j]);
        for (int k0 = 0; k0 < kw; k0 += 64) {
#pragma unroll
            for (int j = 0; j < 4; ++j) b1[j] = load_b_g(pB[j] + 32);
            v16bf a00 = load_a_lds(pA0);
            v16bf a10 = load_a_lds(pA1);
#pragma unroll
            for (int j = 0; j < 4; ++j) acc[0][j] = wmma_bf16(a00, b0[j], acc[0][j]);
#pragma unroll
            for (int j = 0; j < 4; ++j) acc[1][j] = wmma_bf16(a10, b0[j], acc[1][j]);
#pragma unroll
            for (int j = 0; j < 4; ++j) { pB[j] += 64; b0[j] = load_b_g(pB[j]); } // stream next
            v16bf a01 = load_a_lds(pA0 + 32);
            v16bf a11 = load_a_lds(pA1 + 32);
#pragma unroll
            for (int j = 0; j < 4; ++j) acc[0][j] = wmma_bf16(a01, b1[j], acc[0][j]);
#pragma unroll
            for (int j = 0; j < 4; ++j) acc[1][j] = wmma_bf16(a11, b1[j], acc[1][j]);
            pA0 += 64; pA1 += 64;
        }
        __syncthreads();  // panel reuse barrier
    }

    int rbase = (lane >> 4) << 3;
    int col = lane & 15;
#pragma unroll
    for (int mt = 0; mt < 2; ++mt) {
#pragma unroll
        for (int j = 0; j < 4; ++j) {
            int n = n0 + j * 16 + col;
            float bs = bias ? bias[n] : 0.f;
#pragma unroll
            for (int i = 0; i < 8; ++i) {
                int m = m0 + mt * 16 + rbase + i;
                if (m < M) {
                    float v = acc[mt][j][i] + bs;
                    size_t idx = (size_t)m * N + n;
                    if (op == 2) {
                        resid[idx] += v;
                    } else {
                        if (op == 1) v = 0.5f * v * (1.f + erff(v * 0.70710678118f));
                        outh[idx] = (__bf16)v;
                    }
                }
            }
        }
    }
}

// ---------------- QKV GEMM with head-layout scatter ----------------
__global__ __launch_bounds__(128) void gemm_qkv_k(
    const __bf16* __restrict__ A, const __bf16* __restrict__ Wt,
    const float* __restrict__ bias,
    __bf16* __restrict__ Qb, __bf16* __restrict__ Kb, __bf16* __restrict__ Vt)
{
    constexpr int K = D_;
    __shared__ __align__(16) __bf16 sA[32 * K];   // 16 KB
    int wave = threadIdx.x >> 5, lane = threadIdx.x & 31;
    int m0 = blockIdx.x * 32;
    int n0 = blockIdx.y * 256 + wave * 64;
    int rowsA = (M_ - m0 >= 32) ? 32 : (M_ - m0);

    if (wave == 0) tdm_load_panel(sA, A + (size_t)m0 * K, K, rowsA, K);
    __syncthreads();

    v8f acc[2][4];
#pragma unroll
    for (int mt = 0; mt < 2; ++mt)
#pragma unroll
        for (int j = 0; j < 4; ++j) acc[mt][j] = v8f{};

    const __bf16* pB[4];
#pragma unroll
    for (int j = 0; j < 4; ++j)
        pB[j] = b_lane_base(Wt + (size_t)(n0 + j * 16) * K, K);
    const __bf16* pA0 = a_lane_base(sA, K);
    const __bf16* pA1 = pA0 + (size_t)16 * K;

    v16bf b0[4], b1[4];
#pragma unroll
    for (int j = 0; j < 4; ++j) b0[j] = load_b_g(pB[j]);
    for (int k0 = 0; k0 < K; k0 += 64) {
#pragma unroll
        for (int j = 0; j < 4; ++j) b1[j] = load_b_g(pB[j] + 32);
        v16bf a00 = load_a_lds(pA0);
        v16bf a10 = load_a_lds(pA1);
#pragma unroll
        for (int j = 0; j < 4; ++j) acc[0][j] = wmma_bf16(a00, b0[j], acc[0][j]);
#pragma unroll
        for (int j = 0; j < 4; ++j) acc[1][j] = wmma_bf16(a10, b0[j], acc[1][j]);
#pragma unroll
        for (int j = 0; j < 4; ++j) { pB[j] += 64; b0[j] = load_b_g(pB[j]); }
        v16bf a01 = load_a_lds(pA0 + 32);
        v16bf a11 = load_a_lds(pA1 + 32);
#pragma unroll
        for (int j = 0; j < 4; ++j) acc[0][j] = wmma_bf16(a01, b1[j], acc[0][j]);
#pragma unroll
        for (int j = 0; j < 4; ++j) acc[1][j] = wmma_bf16(a11, b1[j], acc[1][j]);
        pA0 += 64; pA1 += 64;
    }

    int rbase = (lane >> 4) << 3;
    int col = lane & 15;
#pragma unroll
    for (int mt = 0; mt < 2; ++mt) {
#pragma unroll
        for (int j = 0; j < 4; ++j) {
            int n = n0 + j * 16 + col;
            float bs = bias[n];
            int s = n >> 8, rem = n & 255, h = rem >> 5, hd = rem & 31;
#pragma unroll
            for (int i = 0; i < 8; ++i) {
                int m = m0 + mt * 16 + rbase + i;
                if (m < M_) {
                    int b = m / T_, t = m - b * T_;
                    __bf16 hv = (__bf16)(acc[mt][j][i] + bs);
                    size_t bh = (size_t)b * H_ + h;
                    if (s == 0)      Qb[(bh * TP_ + t) * HD_ + hd] = hv;
                    else if (s == 1) Kb[(bh * TP_ + t) * HD_ + hd] = hv;
                    else             Vt[(bh * HD_ + hd) * TP_ + t] = hv;
                }
            }
        }
    }
}

// ---------------- masked attention ----------------
__device__ __forceinline__ bool mask_ok(int q, int k, int l0, int l1) {
    constexpr int NSS = L0_ + L1_, NSE = L0_ + L1_ + NS_;
    if (q >= T_ || k >= T_) return false;
    bool kv = (k < L0_) ? (k < l0) : ((k < NSS) ? ((k - L0_) < l1) : true);
    if (!kv) return false;
    if (q < L0_) return (k <= q) || (k >= NSS && k < NSE);
    if (q < NSS) return (k >= L0_ && k <= q) || (k >= NSS && k < NSE);
    if (q < NSE) return (k < NSE);
    return (k < NSE) || (k == T_ - 1);
}

__global__ __launch_bounds__(64) void attn_k(
    const __bf16* __restrict__ Qb, const __bf16* __restrict__ Kb,
    const __bf16* __restrict__ Vt, const int* __restrict__ len0,
    const int* __restrict__ len1, __bf16* __restrict__ AO)
{
    __shared__ __align__(16) __bf16 sc[16 * TP_];
    __shared__ float red[64];
    __shared__ float rowmax[16], rowsum[16];

    int qt = blockIdx.x, h = blockIdx.y, b = blockIdx.z;
    int tid = threadIdx.x, lane = tid & 31, wave = tid >> 5;
    size_t bh = (size_t)b * H_ + h;
    int q0 = qt * 16;
    int l0v = len0[b], l1v = len1[b];

    // ---- phase 1: S = Q K^T * scale, masked, into LDS ----
    v16bf qf = load_a_g(a_lane_base(Qb + (bh * TP_ + q0) * HD_, HD_));
    int rbase = (lane >> 4) << 3;
    int kc = lane & 15;
    const __bf16* pK = b_lane_base(Kb + (bh * TP_ + wave * 16) * HD_, HD_);
    v16bf kcur = load_b_g(pK);
    for (int kt = wave; kt < KT_; kt += 2) {
        pK += 32 * HD_;
        v16bf knxt = load_b_g(pK);              // over-reads past end once; slack ok
        v8f c = v8f{};
        c = wmma_bf16(qf, kcur, c);
        int kcol = kt * 16 + kc;
#pragma unroll
        for (int i = 0; i < 8; ++i) {
            int q = q0 + rbase + i;
            bool ok = mask_ok(q, kcol, l0v, l1v);
            float v = c[i] * SCALE_;
            sc[(rbase + i) * TP_ + kcol] = ok ? (__bf16)v : (__bf16)(-INFINITY);
        }
        kcur = knxt;
    }
    __syncthreads();

    // ---- phase 2: row softmax (4 threads per row; exp computed once) ----
    int row = tid >> 2, sub = tid & 3;
    float mx = -INFINITY;
    for (int k = sub; k < TP_; k += 4) mx = fmaxf(mx, (float)sc[row * TP_ + k]);
    red[tid] = mx;
    __syncthreads();
    if (sub == 0) {
        float m2 = fmaxf(fmaxf(red[tid], red[tid + 1]), fmaxf(red[tid + 2], red[tid + 3]));
        rowmax[row] = (m2 == -INFINITY) ? 0.f : m2;
    }
    __syncthreads();
    float rm = rowmax[row];
    float sum = 0.f;
    for (int k = sub; k < TP_; k += 4) {
        float e = expf((float)sc[row * TP_ + k] - rm);
        sc[row * TP_ + k] = (__bf16)e;
        sum += e;
    }
    red[tid] = sum;
    __syncthreads();
    if (sub == 0) rowsum[row] = red[tid] + red[tid + 1] + red[tid + 2] + red[tid + 3];
    __syncthreads();
    float inv = 1.f / fmaxf(rowsum[row], 1e-30f);
    for (int k = sub; k < TP_; k += 4)
        sc[row * TP_ + k] = (__bf16)((float)sc[row * TP_ + k] * inv);
    __syncthreads();

    // ---- phase 3: O = P V (each wave one 16-wide hd tile; pipelined V) ----
    int nt = wave;
    const __bf16* pV = b_lane_base(Vt + (bh * HD_ + nt * 16) * TP_, TP_);
    const __bf16* pP = a_lane_base(sc, TP_);
    v16bf vcur = load_b_g(pV);
    v8f acc = v8f{};
    for (int k0 = 0; k0 < TP_; k0 += 32) {
        pV += 32;
        v16bf vnxt = load_b_g(pV);              // final over-read lands in ws slack
        v16bf pf = load_a_lds(pP);
        acc = wmma_bf16(pf, vcur, acc);
        vcur = vnxt;
        pP += 32;
    }
#pragma unroll
    for (int i = 0; i < 8; ++i) {
        int t = q0 + rbase + i;
        if (t < T_)
            AO[((size_t)b * T_ + t) * D_ + h * HD_ + nt * 16 + kc] = (__bf16)acc[i];
    }
}

// ---------------- host orchestration ----------------
extern "C" void kernel_launch(void* const* d_in, const int* in_sizes, int n_in,
                              void* d_out, int out_size, void* d_ws, size_t ws_size,
                              hipStream_t stream)
{
    const float* s0   = (const float*)d_in[0];
    const float* s1   = (const float*)d_in[1];
    const int*   len0 = (const int*)  d_in[2];
    const int*   len1 = (const int*)  d_in[3];
    const float* nst  = (const float*)d_in[4];
    const float* dom  = (const float*)d_in[5];
    const float* cls  = (const float*)d_in[6];
    const float* ln1w = (const float*)d_in[7];
    const float* ln1b = (const float*)d_in[8];
    const float* qkvw = (const float*)d_in[9];
    const float* qkvb = (const float*)d_in[10];
    const float* outw = (const float*)d_in[11];
    const float* outb = (const float*)d_in[12];
    const float* ln2w = (const float*)d_in[13];
    const float* ln2b = (const float*)d_in[14];
    const float* f1w  = (const float*)d_in[15];
    const float* f1b  = (const float*)d_in[16];
    const float* f2w  = (const float*)d_in[17];
    const float* f2b  = (const float*)d_in[18];
    const float* flnw = (const float*)d_in[19];
    const float* flnb = (const float*)d_in[20];

    char* ws = (char*)d_ws;
    size_t off = 0;
    auto take = [&](size_t bytes) -> void* {
        off = (off + 255) & ~(size_t)255;
        void* p = ws + off;
        off += bytes;
        return p;
    };
    float*  xf    = (float*) take((size_t)M_ * D_ * 4);
    __bf16* lnout = (__bf16*)take((size_t)M_ * D_ * 2);
    __bf16* hb    = (__bf16*)take((size_t)M_ * HM_ * D_ * 2);
    __bf16* ao    = (__bf16*)take((size_t)M_ * D_ * 2);
    __bf16* Qb    = (__bf16*)take((size_t)B_ * H_ * TP_ * HD_ * 2);
    __bf16* Kb    = (__bf16*)take((size_t)B_ * H_ * TP_ * HD_ * 2);
    __bf16* Vt    = (__bf16*)take((size_t)B_ * H_ * HD_ * TP_ * 2);
    __bf16* wqkvT = (__bf16*)take((size_t)3 * D_ * D_ * 2);
    __bf16* woutT = (__bf16*)take((size_t)D_ * D_ * 2);
    __bf16* wf1T  = (__bf16*)take((size_t)HM_ * D_ * D_ * 2);
    __bf16* wf2T  = (__bf16*)take((size_t)D_ * HM_ * D_ * 2);
    (void)take(8192);                     // streaming-prefetch slack
    (void)ws_size; (void)in_sizes; (void)n_in; (void)out_size;

    embed_k<<<dim3(T_, B_), 256, 0, stream>>>(s0, s1, nst, dom, cls, xf);

    for (int l = 0; l < NL_; ++l) {
        prep_wt_k<<<(3 * D_ * D_ + 255) / 256, 256, 0, stream>>>(
            qkvw + (size_t)l * D_ * 3 * D_, wqkvT, D_, 3 * D_);
        prep_wt_k<<<(D_ * D_ + 255) / 256, 256, 0, stream>>>(
            outw + (size_t)l * D_ * D_, woutT, D_, D_);
        prep_wt_k<<<(HM_ * D_ * D_ + 255) / 256, 256, 0, stream>>>(
            f1w + (size_t)l * D_ * HM_ * D_, wf1T, D_, HM_ * D_);
        prep_wt_k<<<(D_ * HM_ * D_ + 255) / 256, 256, 0, stream>>>(
            f2w + (size_t)l * HM_ * D_ * D_, wf2T, HM_ * D_, D_);

        ln_bf16_k<<<M_, 256, 0, stream>>>(xf, ln1w + (size_t)l * D_, ln1b + (size_t)l * D_, lnout);
        gemm_qkv_k<<<dim3(MT32_, 3), 128, 0, stream>>>(
            lnout, wqkvT, qkvb + (size_t)l * 3 * D_, Qb, Kb, Vt);
        attn_k<<<dim3(QT_, H_, B_), 64, 0, stream>>>(Qb, Kb, Vt, len0, len1, ao);
        gemm_bf16_k<<<dim3(MT32_, 1), 128, 0, stream>>>(
            ao, woutT, outb + (size_t)l * D_, M_, D_, D_, 2, nullptr, xf);
        ln_bf16_k<<<M_, 256, 0, stream>>>(xf, ln2w + (size_t)l * D_, ln2b + (size_t)l * D_, lnout);
        gemm_bf16_k<<<dim3(MT32_, 4), 128, 0, stream>>>(
            lnout, wf1T, f1b + (size_t)l * HM_ * D_, M_, HM_ * D_, D_, 1, hb, nullptr);
        gemm_bf16_k<<<dim3(MT32_, 1), 128, 0, stream>>>(
            hb, wf2T, f2b + (size_t)l * D_, M_, D_, HM_ * D_, 2, nullptr, xf);
    }

    final_ln_k<<<B_, 256, 0, stream>>>(xf, flnw, flnb, (float*)d_out);
}
